// General_Classification_39668317945864
// MI455X (gfx1250) — matrix operand order repaired and checked by reading, not compiled, not run on hardware
//
#include <hip/hip_runtime.h>
#include <hip/hip_bf16.h>

#define EMB   128
#define NCLS  128
#define HWSZ  (512 * 512)
#define ROWS_PER_WG 128
#define THREADS 256

typedef __attribute__((ext_vector_type(16))) __bf16 v16bf;
typedef __attribute__((ext_vector_type(8)))  float  v8f;

__device__ __forceinline__ unsigned short f32_to_bf16_rne(float f) {
  unsigned int u = __float_as_uint(f);
  unsigned int r = 0x7FFFu + ((u >> 16) & 1u);
  u += r;
  return (unsigned short)(u >> 16);
}

__global__ __launch_bounds__(THREADS)
void cls_gather_gemm_softmax(const float* __restrict__ gc,
                             const int*   __restrict__ inds,
                             const float* __restrict__ Wc,
                             const float* __restrict__ bc,
                             float*       __restrict__ out,
                             int K) {
  // LDS: weights (bf16, 32KB) + gathered feature tile (bf16, 32KB)
  __shared__ unsigned short sW[NCLS * EMB];        // sW[n*EMB + c] = W[n][c]
  __shared__ unsigned short sA[ROWS_PER_WG * EMB]; // sA[m*EMB + c] = feat[m][c]

  const int tid   = threadIdx.x;
  const int kBase = blockIdx.x * ROWS_PER_WG;

  // ---- Stage classifier weights into LDS as bf16 (L2-resident, trivial cost)
  #pragma unroll 4
  for (int i = tid; i < NCLS * EMB; i += THREADS) {
    sW[i] = f32_to_bf16_rne(Wc[i]);
  }

  // ---- Gather: 2 threads per row, 64 channels each; stride HWSZ between channels
  {
    int row  = tid >> 1;                 // 0..127
    int cb   = (tid & 1) * 64;           // 0 or 64
    int grow = kBase + row;
    int idx  = inds[grow < K ? grow : (K - 1)];
    const float* src = gc + (size_t)idx;
    #pragma unroll 8
    for (int c = 0; c < 64; ++c) {
      sA[row * EMB + cb + c] = f32_to_bf16_rne(src[(size_t)(cb + c) * HWSZ]);
    }
  }
  __syncthreads();

  // ---- Per-wave WMMA: wave w owns rows [w*16, w*16+16) of this tile
  const int wave  = tid >> 5;
  const int lane  = tid & 31;
  const int hi    = lane >> 4;   // half-wave select
  const int lm    = lane & 15;
  const int mBase = wave * 16;

  v8f acc[8];
  #pragma unroll
  for (int t = 0; t < 8; ++t) acc[t] = (v8f)0.0f;

  #pragma unroll
  for (int ks = 0; ks < 4; ++ks) {          // K-dim: 4 steps of 32 (bf16)
    const int k0 = ks * 32;
    // A fragment: 16x32 bf16, row = mBase+lm, K per ISA layout
    union { v16bf v; unsigned int u[8]; } A;
    #pragma unroll
    for (int v = 0; v < 8; ++v) {
      int kk = k0 + 2 * (v & 3) + 16 * (v >> 2) + 8 * hi;
      A.u[v] = *(const unsigned int*)&sA[(mBase + lm) * EMB + kk];
    }
    #pragma unroll
    for (int t = 0; t < 8; ++t) {           // N-dim: 8 tiles of 16 classes
      // B fragment: 32x16 bf16, col n = t*16+lm, B[k][n] = W[n][k]
      union { v16bf v; unsigned int u[8]; } B;
      const int n = t * 16 + lm;
      #pragma unroll
      for (int v = 0; v < 8; ++v) {
        int kk = k0 + 2 * (v & 3) + 16 * (v >> 2) + 8 * hi;
        B.u[v] = *(const unsigned int*)&sW[n * EMB + kk];
      }
      acc[t] = __builtin_amdgcn_wmma_f32_16x16x32_bf16(
          /*neg_a=*/false, A.v, /*neg_b=*/false, B.v,
          /*c_mod=*/(short)0, acc[t], /*reuse_a=*/false, /*reuse_b=*/false);
    }
  }

  // ---- Bias (per-lane column bias, same for all accumulator rows)
  float bias[8];
  #pragma unroll
  for (int t = 0; t < 8; ++t) bias[t] = bc[t * 16 + lm];

  // ---- Softmax per row: D layout => VGPR r holds row (r + 8*hi),
  //      columns spread over 16 lanes (lm) x 8 tiles (t).
  #pragma unroll
  for (int r = 0; r < 8; ++r) {
    float mx = -INFINITY;
    #pragma unroll
    for (int t = 0; t < 8; ++t) mx = fmaxf(mx, acc[t][r] + bias[t]);
    #pragma unroll
    for (int off = 1; off < 16; off <<= 1)
      mx = fmaxf(mx, __shfl_xor(mx, off, 16));

    float e[8];
    float sum = 0.0f;
    #pragma unroll
    for (int t = 0; t < 8; ++t) {
      e[t] = __expf(acc[t][r] + bias[t] - mx);
      sum += e[t];
    }
    #pragma unroll
    for (int off = 1; off < 16; off <<= 1)
      sum += __shfl_xor(sum, off, 16);

    const float inv = 1.0f / sum;
    const int row  = mBase + r + 8 * hi;
    const int grow = kBase + row;
    if (grow < K) {
      #pragma unroll
      for (int t = 0; t < 8; ++t)
        out[(size_t)grow * NCLS + t * 16 + lm] = e[t] * inv;
    }
  }
}

extern "C" void kernel_launch(void* const* d_in, const int* in_sizes, int n_in,
                              void* d_out, int out_size, void* d_ws, size_t ws_size,
                              hipStream_t stream) {
  const float* gc   = (const float*)d_in[0];  // [1,128,512,512] f32
  const int*   inds = (const int*)  d_in[1];  // [1,K] indices
  const float* Wc   = (const float*)d_in[2];  // [128,128] f32
  const float* bc   = (const float*)d_in[3];  // [128] f32
  float*       out  = (float*)d_out;          // [K,128] f32
  const int K = in_sizes[1];
  const int grid = (K + ROWS_PER_WG - 1) / ROWS_PER_WG;  // 1024 for K=131072
  cls_gather_gemm_softmax<<<grid, THREADS, 0, stream>>>(gc, inds, Wc, bc, out, K);
}
